// Event3DPoseNet_32289564131443
// MI455X (gfx1250) — compile-verified, matrix operand order checked by reading, not run on hardware
//
#include <hip/hip_runtime.h>
#include <cmath>

typedef __bf16 bf16;
typedef __attribute__((ext_vector_type(16))) __bf16 v16bf;
typedef __attribute__((ext_vector_type(8)))  __bf16 v8bf;
typedef __attribute__((ext_vector_type(8)))  float  v8f;

#define LDK 40   // padded LDS K-stride (bf16 elems); 40*2=80B keeps 16B alignment per 8-elem run

__device__ __forceinline__ int imin(int a,int b){ return a<b?a:b; }
__device__ __forceinline__ float geluf(float x){ return 0.5f*x*(1.f+erff(x*0.70710678118f)); }
__device__ __forceinline__ float actf(float x,int a){
  if(a==1) return x>0.f?x:0.f;
  if(a==2) return geluf(x);
  if(a==3) return 1.f/(1.f+expf(-x));
  return x;
}

struct Epi {
  const float* bias;   // may be null
  const float* scale;  // BN gamma (folded), may be null
  const float* shift;  // BN beta
  float sconst;        // rsqrt(1+eps) folded into scale
  const bf16* res;     // residual, NCHW bf16, may be null
  int rcoff, rctot;
  int act;             // 0 none,1 relu,2 gelu,3 sigmoid
};

// Build a 16x32 bf16 fragment per the CDNA5 A/B-matrix layout:
// lanes 0-15 -> rows 0..15 with K runs [g*8, g*8+8) and [16+g*8, ...), g = lane>>4.
__device__ __forceinline__ v16bf frag16(const bf16* base,int lane){
  int row = lane & 15, g = (lane>>4)&1;
  const bf16* p = base + row*LDK + g*8;
  v8bf lo = *(const v8bf*)p;
  v8bf hi = *(const v8bf*)(p+16);
  v16bf r;
#pragma unroll
  for(int i=0;i<8;i++){ r[i]=lo[i]; r[i+8]=hi[i]; }
  return r;
}
__device__ __forceinline__ v8f wmma_bf16(v16bf a,v16bf b,v8f c){
  return __builtin_amdgcn_wmma_f32_16x16x32_bf16(false,a,false,b,(short)0,c,false,false);
}

// ---------------------------------------------------------------------------
// Implicit-GEMM conv: out[img, co, p] = sum_tap sum_ci Wr[tap,co,ci] * X[img,ci,shift(p,tap)]
// Wr is bf16, repacked+zero-padded to [khw][COp][CIp] (COp mult of 64, CIp mult of 32).
// Block: 128 thr = 4 waves; block tile 64(M=co) x 64(N=pixels); each wave owns a
// 32x32 sub-tile (2x2 accumulators -> 4 WMMA per K-step). Weight (A) tile is staged
// LDS-direct with global_load_async_to_lds_b128 (+ s_wait_asynccnt); activation (B)
// tile is a bounds-checked gather (conv halo / ragged K needs zero-fill).
// ---------------------------------------------------------------------------
template<typename OutT>
__global__ __launch_bounds__(128)
void conv_wmma(const bf16* __restrict__ X,int ci_off,int ci_tot,
               const bf16* __restrict__ Wr,int CIp,int tapstride,
               int CI,int CO,
               int Hin,int Win,int Hout,int Wout,int wsh,   // Wout == 1<<wsh
               int khw,int stride,int pad,
               OutT* __restrict__ Y,int c_off,int c_tot,Epi ep)
{
  __shared__ bf16 sA[64*LDK];
  __shared__ bf16 sB[64*LDK];
  const int tid=threadIdx.x, lane=tid&31, wave=tid>>5;
  const int wm=(wave>>1)&1, wn=wave&1;
  const int m0=blockIdx.x*64, n0=blockIdx.y*64, img=blockIdx.z;
  const int HWo=Hout*Wout;
  const size_t HWi=(size_t)Hin*Win;
  const bf16* Xi = X + ((size_t)img*ci_tot + ci_off)*HWi;
  const int kd=(khw==9)?3:1;
  v8f acc[2][2];
#pragma unroll
  for(int a=0;a<2;a++)
#pragma unroll
    for(int b=0;b<2;b++) acc[a][b]=(v8f){0.f,0.f,0.f,0.f,0.f,0.f,0.f,0.f};

  // B staging map: thread -> pixel row bn (0..63), 16 contiguous k at bkq
  const int bn=tid>>1, bkq=(tid&1)*16;
  const int pB=n0+bn;
  const int ohB=(pB<HWo)?(pB>>wsh):0;
  const int owB=(pB<HWo)?(pB&(Wout-1)):0;

  for(int tap=0;tap<khw;++tap){
    const int dh=tap/kd-pad, dw=tap%kd-pad;
    const int ih=ohB*stride+dh, iw=owB*stride+dw;
    const bool pv=(pB<HWo)&&((unsigned)ih<(unsigned)Hin)&&((unsigned)iw<(unsigned)Win);
    const size_t bbase=(size_t)ih*Win+iw;
    const bf16* Wtap=Wr+(size_t)tap*tapstride;
    for(int k0=0;k0<CI;k0+=32){
      // A tile: 64x32 bf16 = 256 b128 chunks; 2 async LDS-direct loads per thread.
#pragma unroll
      for(int j=0;j<2;j++){
        int cch=tid+128*j, row=cch>>2, kq=(cch&3)*8;
        const bf16* gp=Wtap+(size_t)(m0+row)*CIp+(k0+kq);
        unsigned l3=(unsigned)(size_t)&sA[row*LDK+kq];   // LDS offset (low 32 bits of generic ptr)
        asm volatile("global_load_async_to_lds_b128 %0, %1, off"
                     :: "v"(l3), "v"((unsigned long long)(size_t)gp) : "memory");
      }
      // B tile: gather with pad/bounds zero-fill
#pragma unroll
      for(int i=0;i<16;i++){
        int kg=k0+bkq+i; bf16 v=(bf16)0.f;
        if(pv && kg<CI) v=Xi[(size_t)kg*HWi+bbase];
        sB[bn*LDK+bkq+i]=v;
      }
      if(pv && (k0+32)<CI) __builtin_prefetch(Xi+(size_t)(k0+32+bkq)*HWi+bbase,0,3);
      asm volatile("s_wait_asynccnt 0" ::: "memory");
      __syncthreads();
      v16bf a0=frag16(&sA[(wm*32   )*LDK],lane);
      v16bf a1=frag16(&sA[(wm*32+16)*LDK],lane);
      v16bf b0=frag16(&sB[(wn*32   )*LDK],lane);
      v16bf b1=frag16(&sB[(wn*32+16)*LDK],lane);
      acc[0][0]=wmma_bf16(a0,b0,acc[0][0]);
      acc[0][1]=wmma_bf16(a0,b1,acc[0][1]);
      acc[1][0]=wmma_bf16(a1,b0,acc[1][0]);
      acc[1][1]=wmma_bf16(a1,b1,acc[1][1]);
      __syncthreads();
    }
  }
  // Epilogue: C/D layout lane L -> N=L&15, M=((L>>4)*8)+r
  const int nn=lane&15, mmofs=(lane>>4)<<3;
#pragma unroll
  for(int bi=0;bi<2;bi++){
    int n=n0+wn*32+bi*16+nn;
    if(n>=HWo) continue;
#pragma unroll
    for(int ai=0;ai<2;ai++){
      int mb=m0+wm*32+ai*16+mmofs;
#pragma unroll
      for(int r=0;r<8;r++){
        int m=mb+r;
        if(m<CO){
          float f=acc[ai][bi][r];
          if(ep.bias)  f+=ep.bias[m];
          if(ep.scale) f=f*(ep.scale[m]*ep.sconst)+ep.shift[m];
          if(ep.res)   f+=(float)ep.res[((size_t)img*ep.rctot+ep.rcoff+m)*(size_t)HWo+n];
          f=actf(f,ep.act);
          Y[((size_t)img*c_tot+c_off+m)*(size_t)HWo+n]=(OutT)f;
        }
      }
    }
  }
}

// repack fp32 OIHW weights -> bf16 [tap][COp][CIp], zero-padded
__global__ void repack_w(const float* __restrict__ W,int CO,int CI,int khw,int COp,int CIp,
                         bf16* __restrict__ Y,size_t total){
  size_t i=(size_t)blockIdx.x*blockDim.x+threadIdx.x; if(i>=total) return;
  int ci=(int)(i%CIp); size_t t=i/CIp; int co=(int)(t%COp); int tap=(int)(t/COp);
  float v=0.f;
  if(co<CO && ci<CI) v=W[((size_t)co*CI+ci)*khw+tap];
  Y[i]=(bf16)v;
}

// ------------------------------- elementwise ------------------------------
__global__ void cvt_bf(const float* __restrict__ X,bf16* __restrict__ Y,size_t n){
  size_t i=(size_t)blockIdx.x*blockDim.x+threadIdx.x; if(i<n) Y[i]=(bf16)X[i];
}
__global__ void dw3(const bf16* __restrict__ X,const float* __restrict__ W,const float* __restrict__ bias,
                    int C,int Hin,int Win,int Hout,int Wout,int stride,bf16* __restrict__ Y,size_t total){
  size_t i=(size_t)blockIdx.x*blockDim.x+threadIdx.x; if(i>=total) return;
  int HWo=Hout*Wout;
  int pix=(int)(i%HWo); int t=(int)(i/HWo); int c=t%C; int img=t/C;
  int oh=pix/Wout, ow=pix-oh*Wout;
  const bf16* xp=X+((size_t)img*C+c)*Hin*Win;
  float a=bias[c];
#pragma unroll
  for(int kh=0;kh<3;kh++){ int ih=oh*stride+kh-1; if((unsigned)ih>=(unsigned)Hin) continue;
#pragma unroll
    for(int kw=0;kw<3;kw++){ int iw=ow*stride+kw-1; if((unsigned)iw>=(unsigned)Win) continue;
      a+=W[c*9+kh*3+kw]*(float)xp[(size_t)ih*Win+iw]; } }
  Y[i]=(bf16)a;
}
__global__ void mp2(const bf16* __restrict__ X,int H,int W,bf16* __restrict__ Y,size_t total){
  size_t i=(size_t)blockIdx.x*blockDim.x+threadIdx.x; if(i>=total) return;
  int oH=H/2,oW=W/2; int pix=(int)(i%(oH*oW)); size_t t=i/(oH*oW);
  int oy=pix/oW, ox=pix-oy*oW;
  const bf16* xp=X+t*(size_t)H*W+(size_t)(2*oy)*W+2*ox;
  float a=fmaxf(fmaxf((float)xp[0],(float)xp[1]),fmaxf((float)xp[W],(float)xp[W+1]));
  Y[i]=(bf16)a;
}
__global__ void up2_k(const bf16* __restrict__ X,int H,int W,bf16* __restrict__ Y,size_t total){
  size_t i=(size_t)blockIdx.x*blockDim.x+threadIdx.x; if(i>=total) return;
  int oH=2*H,oW=2*W; int pix=(int)(i%(oH*oW)); size_t t=i/(oH*oW);
  int oy=pix/oW, ox=pix-oy*oW;
  float sy=fminf(fmaxf((oy+0.5f)*0.5f-0.5f,0.f),(float)(H-1));
  float sx=fminf(fmaxf((ox+0.5f)*0.5f-0.5f,0.f),(float)(W-1));
  int y0=(int)sy,y1=imin(y0+1,H-1); float fy=sy-y0;
  int x0=(int)sx,x1=imin(x0+1,W-1); float fx=sx-x0;
  const bf16* xp=X+t*(size_t)H*W;
  float v=((float)xp[(size_t)y0*W+x0]*(1.f-fy)+(float)xp[(size_t)y1*W+x0]*fy)*(1.f-fx)
         +((float)xp[(size_t)y0*W+x1]*(1.f-fy)+(float)xp[(size_t)y1*W+x1]*fy)*fx;
  Y[i]=(bf16)v;
}
__global__ void avgp(const bf16* __restrict__ X,int H,int W,int kh,int kw,bf16* __restrict__ Y,size_t total){
  size_t i=(size_t)blockIdx.x*blockDim.x+threadIdx.x; if(i>=total) return;
  int oH=H/kh,oW=W/kw; int pix=(int)(i%(oH*oW)); size_t t=i/(oH*oW);
  int oy=pix/oW, ox=pix-oy*oW;
  const bf16* xp=X+t*(size_t)H*W+(size_t)(oy*kh)*W+ox*kw;
  float s=0.f;
  for(int a=0;a<kh;a++) for(int b=0;b<kw;b++) s+=(float)xp[(size_t)a*W+b];
  Y[i]=(bf16)(s/(float)(kh*kw));
}
__global__ void cpyc(const bf16* __restrict__ S,int Cs,int HW,bf16* __restrict__ D,int coff,int ctot,size_t total){
  size_t i=(size_t)blockIdx.x*blockDim.x+threadIdx.x; if(i>=total) return;
  int pix=(int)(i%HW); int t=(int)(i/HW); int c=t%Cs; int img=t/Cs;
  D[((size_t)img*ctot+coff+c)*HW+pix]=S[i];
}
// per-pixel layernorm across channels (NCHW), one wave32 per pixel
__global__ void ln_chan(const bf16* __restrict__ X,int coff,int ctot,int C,int HW,
                        const float* __restrict__ g,const float* __restrict__ b,
                        bf16* __restrict__ Y,int ycoff,int yctot,int dogelu){
  int p=blockIdx.x, lane=threadIdx.x;
  int img=p/HW, pix=p-img*HW;
  const bf16* xp=X+((size_t)img*ctot+coff)*HW+pix;
  float s=0.f,s2=0.f;
  for(int c=lane;c<C;c+=32){ float v=(float)xp[(size_t)c*HW]; s+=v; s2+=v*v; }
  for(int o=16;o>0;o>>=1){ s+=__shfl_xor(s,o,32); s2+=__shfl_xor(s2,o,32); }
  float m=s/C, var=s2/C-m*m, rs=rsqrtf(var+1e-5f);
  bf16* yp=Y+((size_t)img*yctot+ycoff)*HW+pix;
  for(int c=lane;c<C;c+=32){
    float v=((float)xp[(size_t)c*HW]-m)*rs*g[c]+b[c];
    if(dogelu) v=geluf(v);
    yp[(size_t)c*HW]=(bf16)v;
  }
}
__global__ void gn_gelu(const bf16* __restrict__ X,int C,int groups,int HW,
                        const float* __restrict__ g,const float* __restrict__ b,bf16* __restrict__ Y){
  int img=blockIdx.x, gr=blockIdx.y;
  int Cg=C/groups;
  size_t base=((size_t)img*C+(size_t)gr*Cg)*HW;
  int n=Cg*HW;
  float s=0.f,s2=0.f;
  for(int e=threadIdx.x;e<n;e+=blockDim.x){ float v=(float)X[base+e]; s+=v; s2+=v*v; }
  __shared__ float r0[256],r1[256];
  r0[threadIdx.x]=s; r1[threadIdx.x]=s2; __syncthreads();
  for(int o=blockDim.x>>1;o>0;o>>=1){ if(threadIdx.x<(unsigned)o){ r0[threadIdx.x]+=r0[threadIdx.x+o]; r1[threadIdx.x]+=r1[threadIdx.x+o]; } __syncthreads(); }
  float m=r0[0]/n, var=r1[0]/n-m*m, rs=rsqrtf(var+1e-5f);
  for(int e=threadIdx.x;e<n;e+=blockDim.x){
    int c=gr*Cg+e/HW;
    float v=((float)X[base+e]-m)*rs*g[c]+b[c];
    Y[base+e]=(bf16)geluf(v);
  }
}
// ---- S5 helpers ----
__global__ void abar_k(const float* lam,const float* ls,float* Ab,int P){
  int p=threadIdx.x; if(p<P) Ab[p]=expf(lam[p]*expf(ls[p]));
}
__global__ void bbar_k(const float* lam,const float* Bm,const float* Ab,float* Bb,int D,size_t total){
  size_t i=(size_t)blockIdx.x*blockDim.x+threadIdx.x; if(i>=total) return;
  int p=(int)(i/D);
  Bb[i]=((Ab[p]-1.f)/lam[p])*Bm[i];
}
__global__ void supd(const float* __restrict__ st,const bf16* __restrict__ Bu,const float* __restrict__ Ab,
                     bf16* __restrict__ ns,int C,int HW,size_t total){
  size_t i=(size_t)blockIdx.x*blockDim.x+threadIdx.x; if(i>=total) return;
  int c=(int)((i/HW)%C);
  ns[i]=(bf16)(st[i]*Ab[c]+(float)Bu[i]);
}
__global__ void ycomb(const bf16* __restrict__ yg,const bf16* __restrict__ fx,const float* __restrict__ Dv,
                      bf16* __restrict__ y,int C,int HW,size_t total){
  size_t i=(size_t)blockIdx.x*blockDim.x+threadIdx.x; if(i>=total) return;
  int c=(int)((i/HW)%C);
  y[i]=(bf16)((float)yg[i]+(float)fx[i]*Dv[c]);
}
__global__ void geluadd(const bf16* __restrict__ y,const bf16* __restrict__ fx,bf16* __restrict__ o,size_t total){
  size_t i=(size_t)blockIdx.x*blockDim.x+threadIdx.x; if(i>=total) return;
  o[i]=(bf16)(geluf((float)y[i])+(float)fx[i]);
}
__global__ void gate_k(const bf16* __restrict__ z,bf16* __restrict__ t,int C,int HW,size_t total){
  size_t i=(size_t)blockIdx.x*blockDim.x+threadIdx.x; if(i>=total) return;
  int chw=C*HW; int img=(int)(i/chw); int r=(int)(i%chw);
  const bf16* zb=z+(size_t)img*2*chw;
  float a=(float)zb[r], g=(float)zb[chw+r];
  t[i]=(bf16)(a*geluf(g));
}
// ---- heads/outputs ----
__global__ void sigm(const float* __restrict__ x,float* __restrict__ y,size_t n){
  size_t i=(size_t)blockIdx.x*blockDim.x+threadIdx.x; if(i<n) y[i]=1.f/(1.f+expf(-x[i]));
}
__global__ void resize_ac_k(const float* __restrict__ X,int h,int w,float* __restrict__ Y,int oh,int ow,int B){
  size_t total=(size_t)B*oh*ow;
  size_t i=(size_t)blockIdx.x*blockDim.x+threadIdx.x; if(i>=total) return;
  int pix=(int)(i%((size_t)oh*ow)); int img=(int)(i/((size_t)oh*ow));
  int oy=pix/ow, ox=pix-oy*ow;
  float ys=(float)oy*(float)(h-1)/(float)(oh-1);
  float xs=(float)ox*(float)(w-1)/(float)(ow-1);
  int y0=(int)floorf(ys),y1=imin(y0+1,h-1); float wy=ys-y0;
  int x0=(int)floorf(xs),x1=imin(x0+1,w-1); float wx=xs-x0;
  const float* xp=X+(size_t)img*h*w;
  float v=(xp[(size_t)y0*w+x0]*(1.f-wy)+xp[(size_t)y1*w+x0]*wy)*(1.f-wx)
         +(xp[(size_t)y0*w+x1]*(1.f-wy)+xp[(size_t)y1*w+x1]*wy)*wx;
  Y[i]=v;
}
__global__ void mean_cm(const bf16* __restrict__ X,int C,int HW,int B,bf16* __restrict__ Y){
  int i=blockIdx.x*blockDim.x+threadIdx.x; if(i>=B*C) return;
  int c=i%C, img=i/C;
  const bf16* xp=X+((size_t)img*C+c)*HW;
  float s=0.f; for(int e=0;e<HW;e++) s+=(float)xp[e];
  Y[(size_t)c*B+img]=(bf16)(s/(float)HW);
}
__global__ void pose_t(const float* __restrict__ P,bf16* __restrict__ Y,int B,int F){
  int i=blockIdx.x*blockDim.x+threadIdx.x; if(i>=B*F) return;
  int c=i%F, img=i/F;
  Y[(size_t)c*B+img]=(bf16)P[(size_t)img*F+c];
}
__global__ void fin_delta(const float* __restrict__ m2,const float* __restrict__ prev,
                          float* __restrict__ delta,float* __restrict__ cur,int B,int F){
  int i=blockIdx.x*blockDim.x+threadIdx.x; if(i>=B*F) return;
  int img=i/F, c=i%F;
  float d=m2[(size_t)c*B+img];
  delta[i]=d; cur[i]=prev[i]+d;
}

// ===========================================================================
extern "C" void kernel_launch(void* const* d_in,const int* in_sizes,int n_in,
                              void* d_out,int out_size,void* d_ws,size_t ws_size,
                              hipStream_t stream)
{
  (void)in_sizes;(void)n_in;(void)out_size;(void)ws_size;
  const int B=64;
  // ---- parameter index map (make_params insertion order) ----
  int id=3;
  const int i_conv1_w=id++, i_conv1_b=id++;
  int bbase[4]; const int bn_[4]={3,4,5,6};
  for(int i=0;i<4;i++){ bbase[i]=id; id+=6+4*bn_[i]; }
  const int i_s5=id; id+=11;
  int dbase[2][4];
  for(int nm=0;nm<2;nm++) for(int k=0;k<4;k++){ dbase[nm][k]=id; id+=8; }
  const int i_head=id; id+=6;
  const int i_sp=id; id+=2;
  const int i_dh=id; id+=12;
  const int i_pe=id; id+=4;
  const int i_mlp=id; id+=10;
  auto F=[&](int i)->const float*{ return (const float*)d_in[i]; };
  const float* in_x=F(0); const float* in_states=F(1); const float* in_pose=F(2);

  // ---- workspace bump allocator ----
  size_t off=0;
  auto alB=[&](size_t e)->bf16*{ bf16* p=(bf16*)((char*)d_ws+off); off+=((e*sizeof(bf16))+255)&~(size_t)255; return p; };
  auto alF=[&](size_t e)->float*{ float* p=(float*)((char*)d_ws+off); off+=((e*sizeof(float))+255)&~(size_t)255; return p; };
  auto ew=[&](size_t n)->dim3{ return dim3((unsigned)((n+255)/256)); };

  const float sconst=1.0f/sqrtf(1.0f+1e-5f);
  auto EP=[&](const float* bias,int act)->Epi{
    Epi e; e.bias=bias;e.scale=nullptr;e.shift=nullptr;e.sconst=1.f;e.res=nullptr;e.rcoff=0;e.rctot=0;e.act=act; return e; };
  auto EPbn=[&](const float* bias,const float* g,const float* b,int act)->Epi{
    Epi e=EP(bias,act); e.scale=g; e.shift=b; e.sconst=sconst; return e; };
  // conv wrapper: repack weights -> bf16 [tap][COp][CIp] then launch implicit-GEMM
  auto conv=[&](const bf16* X,int cioff,int citot,const float* Wp,int CI,int CO,
                int Hin,int Win,int Hout,int Wout,int khw,int stride,int pad,
                void* Y,bool f32,int coff,int ctot,int nimg,Epi ep){
    int COp=(CO+63)&~63, CIp=(CI+31)&~31;
    size_t wn=(size_t)khw*COp*CIp;
    bf16* Wr=alB(wn);
    repack_w<<<ew(wn),256,0,stream>>>(Wp,CO,CI,khw,COp,CIp,Wr,wn);
    int wsh=__builtin_ctz((unsigned)Wout);
    dim3 g((unsigned)((CO+63)/64),(unsigned)((Hout*Wout+63)/64),(unsigned)nimg),b(128);
    if(f32) conv_wmma<float><<<g,b,0,stream>>>(X,cioff,citot,Wr,CIp,COp*CIp,CI,CO,Hin,Win,Hout,Wout,wsh,khw,stride,pad,(float*)Y,coff,ctot,ep);
    else    conv_wmma<bf16 ><<<g,b,0,stream>>>(X,cioff,citot,Wr,CIp,COp*CIp,CI,CO,Hin,Win,Hout,Wout,wsh,khw,stride,pad,(bf16*)Y,coff,ctot,ep);
  };

  // ---- persistent buffers ----
  bf16* fm0=alB((size_t)B*32*48*64);
  bf16* fm1=alB((size_t)B*64*24*32);
  bf16* fm2=alB((size_t)B*128*12*16);
  bf16* fm3=alB((size_t)B*192*6*8);
  bf16* s5last=alB((size_t)B*192*48);
  bf16* nsbuf =alB((size_t)B*192*48);
  bf16* segfin=alB((size_t)B*64*48*64);
  bf16* delfin=alB((size_t)B*64*48*64);
  float* Abar=alF(192);
  float* Bbar=alF(192*192);
  const size_t persist=off;

  // ---- stem + backbone ----
  {
    size_t xin_n=(size_t)B*4*192*256;
    bf16* xin=alB(xin_n);
    cvt_bf<<<ew(xin_n),256,0,stream>>>(in_x,xin,xin_n);
    bf16* c1=alB((size_t)B*16*96*128);
    conv(xin,0,4,F(i_conv1_w),4,16,192,256,96,128,9,2,1,c1,false,0,16,B,EP(F(i_conv1_b),1));

    const bf16* bx=c1; int ci=16,Hin=96,Win=128;
    bf16* fms[4]={fm0,fm1,fm2,fm3};
    const int cos[4]={32,64,128,192};
    for(int i=0;i<4;i++){
      int co=cos[i], n=bn_[i], Ho=Hin/2, Wo=Win/2;
      bf16* adw =alB((size_t)B*ci*Ho*Wo);
      bf16* skin=alB((size_t)B*ci*Ho*Wo);
      bf16* sk  =alB((size_t)B*co*Ho*Wo);
      bf16* xa  =alB((size_t)B*co*Ho*Wo);
      bf16* xb  =alB((size_t)B*co*Ho*Wo);
      bf16* hdw =alB((size_t)B*co*Ho*Wo);
      size_t tci=(size_t)B*ci*Ho*Wo, tco=(size_t)B*co*Ho*Wo;
      dw3<<<ew(tci),256,0,stream>>>(bx,F(bbase[i]+0),F(bbase[i]+1),ci,Hin,Win,Ho,Wo,2,adw,tci);
      mp2<<<ew(tci),256,0,stream>>>(bx,Hin,Win,skin,tci);
      conv(skin,0,ci,F(bbase[i]+4),ci,co,Ho,Wo,Ho,Wo,1,1,0,sk,false,0,co,B,EP(F(bbase[i]+5),0));
      { Epi e=EP(F(bbase[i]+3),1); e.res=sk; e.rctot=co;
        conv(adw,0,ci,F(bbase[i]+2),ci,co,Ho,Wo,Ho,Wo,1,1,0,xa,false,0,co,B,e); }
      bf16* cur=xa;
      for(int j=0;j<n;j++){
        bf16* nxt=(j==n-1)?fms[i]:((cur==xa)?xb:xa);
        dw3<<<ew(tco),256,0,stream>>>(cur,F(bbase[i]+6+4*j),F(bbase[i]+7+4*j),co,Ho,Wo,Ho,Wo,1,hdw,tco);
        Epi e=EP(F(bbase[i]+9+4*j),1); e.res=cur; e.rctot=co;
        conv(hdw,0,co,F(bbase[i]+8+4*j),co,co,Ho,Wo,Ho,Wo,1,1,0,nxt,false,0,co,B,e);
        cur=nxt;
      }
      bx=fms[i]; ci=co; Hin=Ho; Win=Wo;
    }
  }
  off=persist;

  // ---- S5 block (tokens == NCHW @ 6x8, D=P=192) ----
  {
    size_t mark=off;
    const float *ln1g=F(i_s5+0),*ln1b=F(i_s5+1),*lam=F(i_s5+2),*Bm=F(i_s5+3),*Cm=F(i_s5+4),
                *Dv=F(i_s5+5),*lstep=F(i_s5+6),*ln2g=F(i_s5+7),*ln2b=F(i_s5+8),
                *encw=F(i_s5+9),*decw=F(i_s5+10);
    abar_k<<<1,192,0,stream>>>(lam,lstep,Abar,192);
    bbar_k<<<ew(192*192),256,0,stream>>>(lam,Bm,Abar,Bbar,192,(size_t)192*192);
    size_t tot=(size_t)B*192*48;
    bf16* fx=alB(tot);
    ln_chan<<<B*48,32,0,stream>>>(fm3,0,192,192,48,ln1g,ln1b,fx,0,192,0);
    bf16* Bu=alB(tot);
    conv(fx,0,192,Bbar,192,192,6,8,6,8,1,1,0,Bu,false,0,192,B,EP(nullptr,0));
    supd<<<ew(tot),256,0,stream>>>(in_states,Bu,Abar,nsbuf,192,48,tot);
    bf16* yg=alB(tot);
    conv(nsbuf,0,192,Cm,192,192,6,8,6,8,1,1,0,yg,false,0,192,B,EP(nullptr,0));
    bf16* yb=alB(tot);
    ycomb<<<ew(tot),256,0,stream>>>(yg,fx,Dv,yb,192,48,tot);
    bf16* g2=alB(tot);
    geluadd<<<ew(tot),256,0,stream>>>(yb,fx,g2,tot);
    bf16* fx2=alB(tot);
    ln_chan<<<B*48,32,0,stream>>>(g2,0,192,192,48,ln2g,ln2b,fx2,0,192,0);
    bf16* z=alB((size_t)B*384*48);
    conv(fx2,0,192,encw,192,384,6,8,6,8,1,1,0,z,false,0,384,B,EP(nullptr,0));
    bf16* tg=alB(tot);
    gate_k<<<ew(tot),256,0,stream>>>(z,tg,192,48,tot);
    Epi ed=EP(nullptr,0); ed.res=fx2; ed.rctot=192;
    conv(tg,0,192,decw,192,192,6,8,6,8,1,1,0,s5last,false,0,192,B,ed);
    off=mark;
  }

  // ---- decoders (seg=0 -> segfin, del=1 -> delfin) ----
  auto decoder=[&](int nm,bf16* finalbuf){
    size_t mark=off;
    const bf16* x=s5last; int Cx=192,Hc=6,Wc=8;
    const int couts[4]={192,128,64,32};
    const int fmch[4]={192,128,64,32};
    const bf16* fmsrc[4]={fm3,fm2,fm1,fm0};
    for(int k=0;k<4;k++){
      int co=couts[k];
      const bf16* xi=x; int Hi=Hc,Wi=Wc;
      if(k>0){
        int Ho2=Hc*2,Wo2=Wc*2;
        size_t tu=(size_t)B*Cx*Ho2*Wo2;
        bf16* up=alB(tu);
        up2_k<<<ew(tu),256,0,stream>>>(x,Hc,Wc,up,tu);
        xi=up; Hi=Ho2; Wi=Wo2;
      }
      bf16* t0=alB((size_t)B*co*Hi*Wi);
      conv(xi,0,Cx,F(dbase[nm][k]+0),Cx,co,Hi,Wi,Hi,Wi,9,1,1,t0,false,0,co,B,
           EPbn(F(dbase[nm][k]+1),F(dbase[nm][k]+2),F(dbase[nm][k]+3),1));
      int ctot=co+fmch[k];
      bf16* cc=(k==3)?finalbuf:alB((size_t)B*ctot*Hi*Wi);
      conv(t0,0,co,F(dbase[nm][k]+4),co,co,Hi,Wi,Hi,Wi,9,1,1,cc,false,0,ctot,B,
           EPbn(F(dbase[nm][k]+5),F(dbase[nm][k]+6),F(dbase[nm][k]+7),1));
      size_t tc=(size_t)B*fmch[k]*Hi*Wi;
      cpyc<<<ew(tc),256,0,stream>>>(fmsrc[k],fmch[k],Hi*Wi,cc,co,ctot,tc);
      x=cc; Cx=ctot; Hc=Hi; Wc=Wi;
    }
    off=mark;
  };
  decoder(0,segfin);
  decoder(1,delfin);

  // ---- output pointers ----
  float* out_delta=(float*)d_out;
  float* out_seg  =out_delta+(size_t)B*39;
  float* out_sfd  =out_seg+(size_t)B*192*256;
  float* out_st   =out_sfd+(size_t)B*48*64;
  float* out_cur  =out_st +(size_t)B*192*48;

  // ---- segmentation head + st update ----
  {
    size_t mark=off;
    bf16* hb=alB((size_t)B*64*48*64);
    conv(segfin,0,64,F(i_head+0),64,64,48,64,48,64,9,1,1,hb,false,0,64,B,
         EPbn(F(i_head+1),F(i_head+2),F(i_head+3),1));
    float* segf=alF((size_t)B*48*64);
    conv(hb,0,64,F(i_head+4),64,1,48,64,48,64,1,1,0,segf,true,0,1,B,EP(F(i_head+5),0));
    sigm<<<ew((size_t)B*48*64),256,0,stream>>>(segf,out_sfd,(size_t)B*48*64);
    resize_ac_k<<<ew((size_t)B*192*256),256,0,stream>>>(segf,48,64,out_seg,192,256,B);
    bf16* pooled=alB((size_t)B*64*48);
    avgp<<<ew((size_t)B*64*48),256,0,stream>>>(segfin,48,64,8,8,pooled,(size_t)B*64*48);
    Epi est=EP(F(i_sp+1),0); est.res=nsbuf; est.rctot=192;
    conv(pooled,0,64,F(i_sp+0),64,192,6,8,6,8,1,1,0,out_st,true,0,192,B,est);
    off=mark;
  }

  // ---- delta head ----
  {
    size_t mark=off;
    bf16* d0 =alB((size_t)B*128*24*32);
    conv(delfin,0,64,F(i_dh+0),64,128,48,64,24,32,9,2,1,d0,false,0,128,B,EP(F(i_dh+1),0));
    bf16* d0g=alB((size_t)B*128*24*32);
    gn_gelu<<<dim3(B,8),256,0,stream>>>(d0,128,8,24*32,F(i_dh+2),F(i_dh+3),d0g);
    bf16* d1 =alB((size_t)B*256*12*16);
    conv(d0g,0,128,F(i_dh+4),128,256,24,32,12,16,9,2,1,d1,false,0,256,B,EP(F(i_dh+5),0));
    bf16* d1g=alB((size_t)B*256*12*16);
    gn_gelu<<<dim3(B,16),256,0,stream>>>(d1,256,16,12*16,F(i_dh+6),F(i_dh+7),d1g);
    bf16* d2 =alB((size_t)B*512*48);
    conv(d1g,0,256,F(i_dh+8),256,512,12,16,6,8,9,2,1,d2,false,0,512,B,EP(F(i_dh+9),0));
    bf16* d2g=alB((size_t)B*512*48);
    gn_gelu<<<dim3(B,32),256,0,stream>>>(d2,512,32,48,F(i_dh+10),F(i_dh+11),d2g);
    // feature matrix [576][B]: rows 0..511 = spatial mean, rows 512..575 = pose embed
    bf16* feat=alB((size_t)576*B);
    mean_cm<<<ew((size_t)B*512),256,0,stream>>>(d2g,512,48,B,feat);
    bf16* poseT=alB((size_t)39*B);
    pose_t<<<ew((size_t)B*39),256,0,stream>>>(in_pose,poseT,B,39);
    bf16* pe0=alB((size_t)64*B);
    conv(poseT,0,39,F(i_pe+0),39,64,1,B,1,B,1,1,0,pe0,false,0,64,1,EP(F(i_pe+1),1));
    conv(pe0,0,64,F(i_pe+2),64,64,1,B,1,B,1,1,0,feat,false,512,576,1,EP(F(i_pe+3),0));
    bf16* h0 =alB((size_t)2048*B);
    conv(feat,0,576,F(i_mlp+0),576,2048,1,B,1,B,1,1,0,h0,false,0,2048,1,EP(F(i_mlp+1),0));
    bf16* h0n=alB((size_t)2048*B);
    ln_chan<<<B,32,0,stream>>>(h0,0,2048,2048,B,F(i_mlp+2),F(i_mlp+3),h0n,0,2048,1);
    bf16* h1 =alB((size_t)512*B);
    conv(h0n,0,2048,F(i_mlp+4),2048,512,1,B,1,B,1,1,0,h1,false,0,512,1,EP(F(i_mlp+5),0));
    bf16* h1n=alB((size_t)512*B);
    ln_chan<<<B,32,0,stream>>>(h1,0,512,512,B,F(i_mlp+6),F(i_mlp+7),h1n,0,512,1);
    float* m2=alF((size_t)39*B);
    conv(h1n,0,512,F(i_mlp+8),512,39,1,B,1,B,1,1,0,m2,true,0,39,1,EP(F(i_mlp+9),0));
    fin_delta<<<ew((size_t)B*39),256,0,stream>>>(m2,in_pose,out_delta,out_cur,B,39);
    off=mark;
  }
}
